// TransformerDecoderCABlock_49271864819776
// MI455X (gfx1250) — matrix operand order repaired and checked by reading, not compiled, not run on hardware
//
#include <hip/hip_runtime.h>
#include <math.h>

// ---------------------------------------------------------------------------
// TransformerDecoder block (self-attn + cross-attn + FFN) for MI455X gfx1250.
// All GEMM-shaped math on v_wmma_f32_16x16x32_bf16 (fp32 accumulate).
// ~103 GFLOP over ~100MB (AI ~ 1000 FLOP/B) -> compute bound; the design goal
// is WMMA occupancy. GEMM: 128x128 tile, double-buffered LDS fed by
// global_load_async_to_lds_b128 (ASYNCcnt). All B operands are pre-transposed
// weights, so staging is pure contiguous 16B copies (no in-kernel transpose).
// Flash attention reads K and V^T fragments straight from global; no barriers.
// ---------------------------------------------------------------------------

typedef __attribute__((ext_vector_type(16))) __bf16 v16bf;
typedef __attribute__((ext_vector_type(8)))  __bf16 v8bf;
typedef __attribute__((ext_vector_type(8)))  float  v8f;
typedef int v4i __attribute__((vector_size(16)));

#define DEV __device__ __forceinline__

static constexpr int Bc = 4, Tt = 1024, Sc = 1024, Ec = 768, Hc = 12, Dc = 64, Mc = 3072;
static constexpr int HD = Hc * Dc;           // 768
static constexpr float SCALE = 0.125f;       // 1/sqrt(64)
static constexpr float EPS = 1e-5f;

DEV __bf16 f2bf(float x) {
  union { float f; unsigned u; } c; c.f = x;
  unsigned r = c.u + 0x7fffu + ((c.u >> 16) & 1u);   // round-to-nearest-even
  unsigned short h = (unsigned short)(r >> 16);
  return __builtin_bit_cast(__bf16, h);
}

DEV v8f zero8() {
  v8f r;
  #pragma unroll
  for (int i = 0; i < 8; ++i) r[i] = 0.f;
  return r;
}

DEV v16bf pack16(v8bf lo, v8bf hi) {
  v16bf r;
  #pragma unroll
  for (int i = 0; i < 8; ++i) { r[i] = lo[i]; r[8 + i] = hi[i]; }
  return r;
}

DEV v8f wmma_bf16(v16bf a, v16bf b, v8f c) {
  return __builtin_amdgcn_wmma_f32_16x16x32_bf16(false, a, false, b, (short)0, c,
                                                 false, false);
}

DEV float gelu_tanh(float x) {
  const float c0 = 0.7978845608028654f, c1 = 0.044715f;
  return 0.5f * x * (1.f + tanhf(c0 * (x + c1 * x * x * x)));
}

// 16-byte global -> LDS copy: async DMA (ASYNCcnt) when available.
#if __has_builtin(__builtin_amdgcn_global_load_async_to_lds_b128)
#define ASYNC_COPY 1
DEV void cp16_g2l(const __bf16* g, __bf16* l) {
  __builtin_amdgcn_global_load_async_to_lds_b128(
      (__attribute__((address_space(1))) v4i*)g,
      (__attribute__((address_space(3))) v4i*)l, 0, 0);
}
DEV void wait_async0() { asm volatile("s_wait_asynccnt 0" ::: "memory"); }
#else
#define ASYNC_COPY 0
DEV void cp16_g2l(const __bf16* g, __bf16* l) { *(v8bf*)l = *(const v8bf*)g; }
DEV void wait_async0() {}
#endif

// --------------------------- elementwise helpers ---------------------------

__global__ void cvt_f32_bf16(const float* __restrict__ in, __bf16* __restrict__ out,
                             size_t n) {
  size_t i = (size_t)blockIdx.x * blockDim.x + threadIdx.x;
  if (i < n) out[i] = f2bf(in[i]);
}

// in [K,N] fp32 -> out [N,K] bf16  (B^T for WMMA GEMM B operands)
__global__ void cvt_transpose_bf16(const float* __restrict__ in,
                                   __bf16* __restrict__ out, int K, int N) {
  size_t idx = (size_t)blockIdx.x * blockDim.x + threadIdx.x;
  if (idx >= (size_t)K * N) return;
  int k = (int)(idx / N), n = (int)(idx % N);
  out[(size_t)n * K + k] = f2bf(in[idx]);
}

// w[h][e][d] fp32 -> out[(h*D+d)][e] bf16 (B^T: N = H*D, K = E)
__global__ void repack_head_wT(const float* __restrict__ w, __bf16* __restrict__ out) {
  int idx = blockIdx.x * blockDim.x + threadIdx.x;          // < H*E*D
  int d = idx % Dc;
  int e = (idx / Dc) % Ec;
  int h = idx / (Dc * Ec);
  out[(size_t)(h * Dc + d) * Ec + e] = f2bf(w[idx]);
}

// one block per row; fp32 in, bf16 out (feeds WMMA GEMMs)
__global__ __launch_bounds__(256) void layernorm_bf16(
    const float* __restrict__ x, const float* __restrict__ g,
    const float* __restrict__ b, __bf16* __restrict__ out, int E) {
  int row = blockIdx.x, t = threadIdx.x;
  int wave = t >> 5, lane = t & 31;
  __shared__ float red[2][8];
  float s = 0.f, s2 = 0.f;
  for (int i = t; i < E; i += 256) {
    float v = x[(size_t)row * E + i];
    s += v; s2 += v * v;
  }
  #pragma unroll
  for (int m = 16; m >= 1; m >>= 1) { s += __shfl_xor(s, m); s2 += __shfl_xor(s2, m); }
  if (lane == 0) { red[0][wave] = s; red[1][wave] = s2; }
  __syncthreads();
  float ts = 0.f, ts2 = 0.f;
  #pragma unroll
  for (int w = 0; w < 8; ++w) { ts += red[0][w]; ts2 += red[1][w]; }
  float mu  = ts / E;
  float var = ts2 / E - mu * mu;
  float inv = rsqrtf(var + EPS);
  for (int i = t; i < E; i += 256) {
    float v = (x[(size_t)row * E + i] - mu) * inv * g[i] + b[i];
    out[(size_t)row * E + i] = f2bf(v);
  }
}

// ------------------------------- WMMA GEMM ---------------------------------
// C[M,N] = A[M,K](bf16,row-major) * Bt[N,K](bf16, pre-transposed B).
// (+bias)(+gelu)(+resid). 128x128 block tile, 8 waves (2x4), each wave 64x32.
// Double-buffered LDS, async global->LDS staging.
__global__ __launch_bounds__(256) void gemm_wmma_bf16(
    const __bf16* __restrict__ A, const __bf16* __restrict__ Bt,
    int Mdim, int Ndim, int Kdim,
    const float* __restrict__ bias, const float* __restrict__ resid,
    float* __restrict__ outF, __bf16* __restrict__ outH,
    __bf16* __restrict__ outHT, int gelu) {
  __shared__ __align__(16) __bf16 As[2][128][40];   // [m][k], padded
  __shared__ __align__(16) __bf16 Bs[2][128][40];   // [n][k], padded

  const int t = threadIdx.x;
  const int wave = t >> 5, lane = t & 31;
  const int wr = wave >> 2, wc = wave & 3;          // wave grid 2x4
  const int hlf = lane >> 4, l16 = lane & 15;
  const int m0 = blockIdx.y * 128, n0 = blockIdx.x * 128;

  v8f acc[4][2];
  #pragma unroll
  for (int a = 0; a < 4; ++a)
    #pragma unroll
    for (int bb = 0; bb < 2; ++bb) acc[a][bb] = zero8();

  auto stage = [&](int buf, int k0) {
    #pragma unroll
    for (int c = t; c < 512; c += 256) {            // A tile 128x32
      int row = c >> 2, col8 = (c & 3) * 8;
      cp16_g2l(A + (size_t)(m0 + row) * Kdim + k0 + col8, &As[buf][row][col8]);
    }
    #pragma unroll
    for (int c = t; c < 512; c += 256) {            // B^T tile 128x32
      int row = c >> 2, col8 = (c & 3) * 8;
      cp16_g2l(Bt + (size_t)(n0 + row) * Kdim + k0 + col8, &Bs[buf][row][col8]);
    }
  };

  const int KT = Kdim / 32;
  int buf = 0;
  stage(0, 0);
  for (int kt = 0; kt < KT; ++kt) {
    wait_async0();
    __syncthreads();
    if (kt + 1 < KT) stage(buf ^ 1, (kt + 1) * 32);

    v16bf af[4], bfr[2];
    #pragma unroll
    for (int a = 0; a < 4; ++a) {
      int row = wr * 64 + a * 16 + l16;
      v8bf lo = *(const v8bf*)(&As[buf][row][hlf * 8]);
      v8bf hi = *(const v8bf*)(&As[buf][row][16 + hlf * 8]);
      af[a] = pack16(lo, hi);
    }
    #pragma unroll
    for (int bb = 0; bb < 2; ++bb) {
      int n = wc * 32 + bb * 16 + l16;
      v8bf lo = *(const v8bf*)(&Bs[buf][n][hlf * 8]);
      v8bf hi = *(const v8bf*)(&Bs[buf][n][16 + hlf * 8]);
      bfr[bb] = pack16(lo, hi);
    }
    #pragma unroll
    for (int a = 0; a < 4; ++a)
      #pragma unroll
      for (int bb = 0; bb < 2; ++bb)
        acc[a][bb] = wmma_bf16(af[a], bfr[bb], acc[a][bb]);
    buf ^= 1;
  }

  // epilogue: C layout -> row = r + 8*hlf (per 16-tile), col = l16
  #pragma unroll
  for (int a = 0; a < 4; ++a)
    #pragma unroll
    for (int bb = 0; bb < 2; ++bb)
      #pragma unroll
      for (int r = 0; r < 8; ++r) {
        int row = m0 + wr * 64 + a * 16 + r + 8 * hlf;
        int col = n0 + wc * 32 + bb * 16 + l16;
        float v = acc[a][bb][r];
        if (bias)  v += bias[col];
        if (gelu)  v = gelu_tanh(v);
        if (resid) v += resid[(size_t)row * Ndim + col];
        if (outF)  outF[(size_t)row * Ndim + col] = v;
        if (outH)  outH[(size_t)row * Ndim + col] = f2bf(v);
        if (outHT) outHT[(size_t)col * Mdim + row] = f2bf(v);  // transposed copy
      }
}

// --------------------------- fused flash attention -------------------------
// Q/K bf16 [B*T, H*D]; V^T bf16 [H*D, B*Tk]. One block = 128 q rows of one
// (b,h); 8 waves, each wave independently owns 16 q rows (no block barriers).
// Keys in 32-wide tiles; online softmax; WMMA for Q*K^T and P*V. P tile goes
// C-layout -> A-layout via a wave-private LDS round trip (s_wait_dscnt 0).
__global__ __launch_bounds__(256) void flash_attn_bf16(
    const __bf16* __restrict__ Qh, const __bf16* __restrict__ Kh,
    const __bf16* __restrict__ VhT, __bf16* __restrict__ Oh,
    int Tq, int Tk, int causal) {
  __shared__ __align__(16) __bf16 Pbuf[8][16][40];   // per-wave P round-trip

  const int t = threadIdx.x;
  const int wave = t >> 5, lane = t & 31;
  const int hlf = lane >> 4, l16 = lane & 15;
  const int b = blockIdx.x / Hc, h = blockIdx.x % Hc;
  const int qbase = blockIdx.y * 128 + wave * 16;
  const size_t NR = (size_t)Bc * Tk;                 // rows of V^T columns

  // preload Q fragments (A operand, 16x64 split into two K=32 chunks)
  const __bf16* qrow = Qh + ((size_t)(b * Tq + qbase + l16)) * HD + h * Dc;
  v16bf qf[2];
  #pragma unroll
  for (int c = 0; c < 2; ++c) {
    v8bf lo = *(const v8bf*)(qrow + c * 32 + hlf * 8);
    v8bf hi = *(const v8bf*)(qrow + c * 32 + 16 + hlf * 8);
    qf[c] = pack16(lo, hi);
  }

  float mrun[8], lrun[8];
  #pragma unroll
  for (int r = 0; r < 8; ++r) { mrun[r] = -1e30f; lrun[r] = 0.f; }
  v8f oacc[4];
  #pragma unroll
  for (int d = 0; d < 4; ++d) oacc[d] = zero8();

  const int ktiles = causal ? ((qbase + 47) >> 5) : (Tk / 32);
  for (int kt = 0; kt < ktiles; ++kt) {
    const int kb = kt * 32;

    // S = Q * K^T  (two 16-key subtiles)
    v8f s[2];
    #pragma unroll
    for (int bb = 0; bb < 2; ++bb) {
      s[bb] = zero8();
      const __bf16* krow =
          Kh + (size_t)(b * Tk + kb + bb * 16 + l16) * HD + h * Dc;
      #pragma unroll
      for (int c = 0; c < 2; ++c) {
        v8bf lo = *(const v8bf*)(krow + c * 32 + hlf * 8);
        v8bf hi = *(const v8bf*)(krow + c * 32 + 16 + hlf * 8);
        s[bb] = wmma_bf16(qf[c], pack16(lo, hi), s[bb]);
      }
    }

    // scale + causal mask
    #pragma unroll
    for (int bb = 0; bb < 2; ++bb)
      #pragma unroll
      for (int r = 0; r < 8; ++r) {
        float sv = s[bb][r] * SCALE;
        int kidx = kb + bb * 16 + l16;
        int qidx = qbase + r + 8 * hlf;
        if (causal && kidx > qidx) sv = -1e30f;
        s[bb][r] = sv;
      }

    // online softmax: row reductions across the 16-lane N group
    float tmax[8], corr[8];
    #pragma unroll
    for (int r = 0; r < 8; ++r) tmax[r] = fmaxf(s[0][r], s[1][r]);
    #pragma unroll
    for (int m = 1; m <= 8; m <<= 1)
      #pragma unroll
      for (int r = 0; r < 8; ++r) tmax[r] = fmaxf(tmax[r], __shfl_xor(tmax[r], m));
    #pragma unroll
    for (int r = 0; r < 8; ++r) {
      float mnew = fmaxf(mrun[r], tmax[r]);
      corr[r] = __expf(mrun[r] - mnew);
      mrun[r] = mnew;
      lrun[r] *= corr[r];
    }
    #pragma unroll
    for (int bb = 0; bb < 2; ++bb)
      #pragma unroll
      for (int r = 0; r < 8; ++r) s[bb][r] = __expf(s[bb][r] - mrun[r]);
    float tsum[8];
    #pragma unroll
    for (int r = 0; r < 8; ++r) tsum[r] = s[0][r] + s[1][r];
    #pragma unroll
    for (int m = 1; m <= 8; m <<= 1)
      #pragma unroll
      for (int r = 0; r < 8; ++r) tsum[r] += __shfl_xor(tsum[r], m);
    #pragma unroll
    for (int r = 0; r < 8; ++r) lrun[r] += tsum[r];
    #pragma unroll
    for (int d = 0; d < 4; ++d)
      #pragma unroll
      for (int r = 0; r < 8; ++r) oacc[d][r] *= corr[r];

    // P: C-layout -> A-layout via wave-private LDS round trip
    #pragma unroll
    for (int bb = 0; bb < 2; ++bb)
      #pragma unroll
      for (int r = 0; r < 8; ++r)
        Pbuf[wave][r + 8 * hlf][bb * 16 + l16] = f2bf(s[bb][r]);
    asm volatile("s_wait_dscnt 0" ::: "memory");   // wave-local LDS RAW fence
    v8bf plo = *(const v8bf*)(&Pbuf[wave][l16][hlf * 8]);
    v8bf phi = *(const v8bf*)(&Pbuf[wave][l16][16 + hlf * 8]);
    v16bf pf = pack16(plo, phi);

    // O += P * V : V^T columns are contiguous in memory -> direct b128 loads
    #pragma unroll
    for (int d = 0; d < 4; ++d) {
      const __bf16* vcol = VhT + (size_t)(h * Dc + d * 16 + l16) * NR +
                           (size_t)b * Tk + kb + hlf * 8;
      v8bf lo = *(const v8bf*)(vcol);
      v8bf hi = *(const v8bf*)(vcol + 16);
      oacc[d] = wmma_bf16(pf, pack16(lo, hi), oacc[d]);
    }
  }

  // normalize + write bf16
  #pragma unroll
  for (int d = 0; d < 4; ++d)
    #pragma unroll
    for (int r = 0; r < 8; ++r) {
      int qidx = qbase + r + 8 * hlf;
      int col = h * Dc + d * 16 + l16;
      float o = oacc[d][r] / lrun[r];
      Oh[(size_t)(b * Tq + qidx) * HD + col] = f2bf(o);
    }
}

// ------------------------------ host driver --------------------------------

extern "C" void kernel_launch(void* const* d_in, const int* in_sizes, int n_in,
                              void* d_out, int out_size, void* d_ws, size_t ws_size,
                              hipStream_t stream) {
  (void)in_sizes; (void)n_in; (void)out_size; (void)ws_size;
  const float* target = (const float*)d_in[0];
  const float* memory = (const float*)d_in[1];
  const float* sa_wq = (const float*)d_in[2];  const float* sa_bq = (const float*)d_in[3];
  const float* sa_wk = (const float*)d_in[4];  const float* sa_bk = (const float*)d_in[5];
  const float* sa_wv = (const float*)d_in[6];  const float* sa_bv = (const float*)d_in[7];
  const float* sa_wo = (const float*)d_in[8];  const float* sa_bo = (const float*)d_in[9];
  const float* ca_wq = (const float*)d_in[10]; const float* ca_bq = (const float*)d_in[11];
  const float* ca_wk = (const float*)d_in[12]; const float* ca_bk = (const float*)d_in[13];
  const float* ca_wv = (const float*)d_in[14]; const float* ca_bv = (const float*)d_in[15];
  const float* ca_wo = (const float*)d_in[16]; const float* ca_bo = (const float*)d_in[17];
  const float* ln1_g = (const float*)d_in[18]; const float* ln1_b = (const float*)d_in[19];
  const float* ln2_g = (const float*)d_in[20]; const float* ln2_b = (const float*)d_in[21];
  const float* ln3_g = (const float*)d_in[22]; const float* ln3_b = (const float*)d_in[23];
  const float* ffn_w1 = (const float*)d_in[24]; const float* ffn_b1 = (const float*)d_in[25];
  const float* ffn_w2 = (const float*)d_in[26]; const float* ffn_b2 = (const float*)d_in[27];

  const int NT = Bc * Tt, NS = Bc * Sc;
  char* ws = (char*)d_ws;
  size_t off = 0;
  auto alloc = [&](size_t bytes) -> char* {
    char* p = ws + off;
    off = (off + bytes + 255) & ~(size_t)255;
    return p;
  };
  // bf16 weights, all stored as B^T = [N][K]
  __bf16* wsaq = (__bf16*)alloc((size_t)HD * Ec * 2);
  __bf16* wsak = (__bf16*)alloc((size_t)HD * Ec * 2);
  __bf16* wsav = (__bf16*)alloc((size_t)HD * Ec * 2);
  __bf16* wcaq = (__bf16*)alloc((size_t)HD * Ec * 2);
  __bf16* wcak = (__bf16*)alloc((size_t)HD * Ec * 2);
  __bf16* wcav = (__bf16*)alloc((size_t)HD * Ec * 2);
  __bf16* wsao = (__bf16*)alloc((size_t)Ec * HD * 2);
  __bf16* wcao = (__bf16*)alloc((size_t)Ec * HD * 2);
  __bf16* wff1 = (__bf16*)alloc((size_t)Mc * Ec * 2);
  __bf16* wff2 = (__bf16*)alloc((size_t)Ec * Mc * 2);
  // activations
  __bf16* memh  = (__bf16*)alloc((size_t)NS * Ec * 2);
  __bf16* xlnh  = (__bf16*)alloc((size_t)NT * Ec * 2);
  __bf16* Qh    = (__bf16*)alloc((size_t)NT * HD * 2);
  __bf16* Kh    = (__bf16*)alloc((size_t)NS * HD * 2);
  __bf16* VhT   = (__bf16*)alloc((size_t)HD * NS * 2);
  __bf16* attnh = (__bf16*)alloc((size_t)NT * HD * 2);
  __bf16* ffh   = (__bf16*)alloc((size_t)NT * Mc * 2);
  float*  x1    = (float*)alloc((size_t)NT * Ec * 4);
  float*  x2    = (float*)alloc((size_t)NT * Ec * 4);

  const int rp = Hc * Ec * Dc;  // 589824, multiple of 256
  repack_head_wT<<<rp / 256, 256, 0, stream>>>(sa_wq, wsaq);
  repack_head_wT<<<rp / 256, 256, 0, stream>>>(sa_wk, wsak);
  repack_head_wT<<<rp / 256, 256, 0, stream>>>(sa_wv, wsav);
  repack_head_wT<<<rp / 256, 256, 0, stream>>>(ca_wq, wcaq);
  repack_head_wT<<<rp / 256, 256, 0, stream>>>(ca_wk, wcak);
  repack_head_wT<<<rp / 256, 256, 0, stream>>>(ca_wv, wcav);
  auto cvtT = [&](const float* src, __bf16* dst, int K, int N) {
    size_t n = (size_t)K * N;
    cvt_transpose_bf16<<<(unsigned)((n + 255) / 256), 256, 0, stream>>>(src, dst, K, N);
  };
  cvtT(sa_wo, wsao, HD, Ec);     // -> [E][HD]
  cvtT(ca_wo, wcao, HD, Ec);
  cvtT(ffn_w1, wff1, Ec, Mc);    // -> [M][E]
  cvtT(ffn_w2, wff2, Mc, Ec);    // -> [E][M]
  {
    size_t n = (size_t)NS * Ec;
    cvt_f32_bf16<<<(unsigned)((n + 255) / 256), 256, 0, stream>>>(memory, memh, n);
  }

  dim3 blk(256);
  dim3 gE(Ec / 128, NT / 128);     // N=768 tiles x M=4096 tiles
  dim3 gM(Mc / 128, NT / 128);
  dim3 gAtt(Bc * Hc, Tt / 128);

  // ---- self attention ----
  layernorm_bf16<<<NT, blk, 0, stream>>>(target, ln1_g, ln1_b, xlnh, Ec);
  gemm_wmma_bf16<<<gE, blk, 0, stream>>>(xlnh, wsaq, NT, HD, Ec, sa_bq, nullptr,
                                         nullptr, Qh, nullptr, 0);
  gemm_wmma_bf16<<<gE, blk, 0, stream>>>(xlnh, wsak, NT, HD, Ec, sa_bk, nullptr,
                                         nullptr, Kh, nullptr, 0);
  gemm_wmma_bf16<<<gE, blk, 0, stream>>>(xlnh, wsav, NT, HD, Ec, sa_bv, nullptr,
                                         nullptr, nullptr, VhT, 0);
  flash_attn_bf16<<<gAtt, blk, 0, stream>>>(Qh, Kh, VhT, attnh, Tt, Tt, 1);
  gemm_wmma_bf16<<<gE, blk, 0, stream>>>(attnh, wsao, NT, Ec, HD, sa_bo, target,
                                         x1, nullptr, nullptr, 0);

  // ---- cross attention ----
  layernorm_bf16<<<NT, blk, 0, stream>>>(x1, ln2_g, ln2_b, xlnh, Ec);
  gemm_wmma_bf16<<<gE, blk, 0, stream>>>(xlnh, wcaq, NT, HD, Ec, ca_bq, nullptr,
                                         nullptr, Qh, nullptr, 0);
  gemm_wmma_bf16<<<gE, blk, 0, stream>>>(memh, wcak, NS, HD, Ec, ca_bk, nullptr,
                                         nullptr, Kh, nullptr, 0);
  gemm_wmma_bf16<<<gE, blk, 0, stream>>>(memh, wcav, NS, HD, Ec, ca_bv, nullptr,
                                         nullptr, nullptr, VhT, 0);
  flash_attn_bf16<<<gAtt, blk, 0, stream>>>(Qh, Kh, VhT, attnh, Tt, Sc, 0);
  gemm_wmma_bf16<<<gE, blk, 0, stream>>>(attnh, wcao, NT, Ec, HD, ca_bo, x1,
                                         x2, nullptr, nullptr, 0);

  // ---- FFN ----
  layernorm_bf16<<<NT, blk, 0, stream>>>(x2, ln3_g, ln3_b, xlnh, Ec);
  gemm_wmma_bf16<<<gM, blk, 0, stream>>>(xlnh, wff1, NT, Mc, Ec, ffn_b1, nullptr,
                                         nullptr, ffh, nullptr, 1);
  gemm_wmma_bf16<<<gE, blk, 0, stream>>>(ffh, wff2, NT, Ec, Mc, ffn_b2, x2,
                                         (float*)d_out, nullptr, nullptr, 0);
}
